// EncoderStack_26929444946349
// MI455X (gfx1250) — compile-verified, matrix operand order checked by reading, not run on hardware
//
#include <hip/hip_runtime.h>
#include <math.h>

// ---------------------------------------------------------------------------
// Types for CDNA5 WMMA (wave32): v16 bf16 operands, v8 f32 accumulator.
// ---------------------------------------------------------------------------
typedef __bf16        bf16x16 __attribute__((ext_vector_type(16)));
typedef float         f32x8   __attribute__((ext_vector_type(8)));
typedef unsigned short us8    __attribute__((ext_vector_type(8)));

union BFrag { bf16x16 v; us8 h[2]; };

static __device__ __forceinline__ unsigned short f2bf(float f) {
    unsigned int u = __float_as_uint(f);
    u += 0x7FFFu + ((u >> 16) & 1u);   // round-to-nearest-even
    return (unsigned short)(u >> 16);
}

// ---------------------------------------------------------------------------
// f32 -> bf16 conversion (grid-stride)
// ---------------------------------------------------------------------------
__global__ void cvt_f32_bf16(const float* __restrict__ in,
                             unsigned short* __restrict__ out, long long n) {
    long long i = (long long)blockIdx.x * blockDim.x + threadIdx.x;
    long long stride = (long long)gridDim.x * blockDim.x;
    for (; i < n; i += stride) out[i] = f2bf(in[i]);
}

// ---------------------------------------------------------------------------
// Batched bf16 WMMA GEMM:  C = scale * (A @ B[^T]) (+bias) (+gelu)
//   A: [M,K] bf16 row-major (lda).  TRANSB=0: B [K,N] (ldb=N);
//   TRANSB=1: B [N,K] (ldb=K), computes A@B^T.
//   Shapes must be multiples of 128(M) x 128(N) x 32(K) -- all ours are.
// Block: 256 thr = 8 wave32 waves (4x2); block tile 128x128;
// wave tile 32x64 -> 8 WMMAs per 32-K step, A fragments reused 4x.
// ---------------------------------------------------------------------------
template <int TRANSB>
__global__ __launch_bounds__(256) void wmma_gemm_bf16(
    const unsigned short* A, const unsigned short* Bm,
    float* outF, unsigned short* outB, const float* bias,
    int K, int lda, int ldb, int ldc,
    long long sA, long long sB, long long sC,
    float scale, int doGelu)
{
    __shared__ unsigned short Ash[128 * 32];  // [m][k], k contiguous
    __shared__ unsigned short Bsh[128 * 32];  // [n][k], k contiguous

    const int tid  = threadIdx.x;
    const int lane = tid & 31;                // wave32
    const int wid  = tid >> 5;                // 0..7
    const int wm   = wid & 3;                 // wave m-tile (32 rows each)
    const int wn   = wid >> 2;                // wave n-half (64 cols each)

    const long long zb = blockIdx.z;
    A  += zb * sA;
    Bm += zb * sB;
    const long long cbase = zb * sC;

    const int row0 = blockIdx.y * 128;
    const int col0 = blockIdx.x * 128;

    f32x8 acc[2][4];
    for (int m = 0; m < 2; ++m)
        for (int t = 0; t < 4; ++t)
            for (int i = 0; i < 8; ++i) acc[m][t][i] = 0.f;

    // global->LDS load coordinates (128x32 tiles, 16 elems/thread)
    const int grow_ = tid >> 1;               // 0..127
    const int gcol_ = (tid & 1) * 16;         // 0 or 16
    const int brow_n = tid >> 3;              // NN: 0..31  (k)
    const int bcol_n = (tid & 7) * 16;        // NN: n offset

    // fragment gather coords (16-bit operand layout, wave32):
    // lanes 0-15 hold K=0-7 & 16-23; lanes 16-31 hold K=8-15 & 24-31
    const int frow = lane & 15;
    const int kb   = (lane >> 4) * 8;

    for (int k0 = 0; k0 < K; k0 += 32) {
        // A tile: rows row0..row0+127, cols k0..k0+31
        *(us8*)&Ash[grow_ * 32 + gcol_] =
            *(const us8*)&A[(long long)(row0 + grow_) * lda + k0 + gcol_];
        *(us8*)&Ash[grow_ * 32 + gcol_ + 8] =
            *(const us8*)&A[(long long)(row0 + grow_) * lda + k0 + gcol_ + 8];
        if (TRANSB) {
            // B rows are n, cols are k: direct copy into [n][k]
            *(us8*)&Bsh[grow_ * 32 + gcol_] =
                *(const us8*)&Bm[(long long)(col0 + grow_) * ldb + k0 + gcol_];
            *(us8*)&Bsh[grow_ * 32 + gcol_ + 8] =
                *(const us8*)&Bm[(long long)(col0 + grow_) * ldb + k0 + gcol_ + 8];
        } else {
            // B rows are k, cols are n: transpose into [n][k]
            us8 v0 = *(const us8*)&Bm[(long long)(k0 + brow_n) * ldb + col0 + bcol_n];
            us8 v1 = *(const us8*)&Bm[(long long)(k0 + brow_n) * ldb + col0 + bcol_n + 8];
#pragma unroll
            for (int i = 0; i < 8; ++i) {
                Bsh[(bcol_n + i)     * 32 + brow_n] = v0[i];
                Bsh[(bcol_n + 8 + i) * 32 + brow_n] = v1[i];
            }
        }
        __syncthreads();

        BFrag a0, a1;
        a0.h[0] = *(const us8*)&Ash[(wm * 32 + frow) * 32 + kb];
        a0.h[1] = *(const us8*)&Ash[(wm * 32 + frow) * 32 + kb + 16];
        a1.h[0] = *(const us8*)&Ash[(wm * 32 + 16 + frow) * 32 + kb];
        a1.h[1] = *(const us8*)&Ash[(wm * 32 + 16 + frow) * 32 + kb + 16];
#pragma unroll
        for (int t = 0; t < 4; ++t) {
            const int nc = wn * 64 + t * 16 + frow;
            BFrag b;
            b.h[0] = *(const us8*)&Bsh[nc * 32 + kb];
            b.h[1] = *(const us8*)&Bsh[nc * 32 + kb + 16];
            acc[0][t] = __builtin_amdgcn_wmma_f32_16x16x32_bf16(
                false, a0.v, false, b.v, (short)0, acc[0][t], false, false);
            acc[1][t] = __builtin_amdgcn_wmma_f32_16x16x32_bf16(
                false, a1.v, false, b.v, (short)0, acc[1][t], false, false);
        }
        __syncthreads();
    }

    // epilogue: C layout -> lane n = lane&15, row = r + 8*(lane>>4)
    const int rhi = (lane >> 4) * 8;
#pragma unroll
    for (int m = 0; m < 2; ++m) {
#pragma unroll
        for (int t = 0; t < 4; ++t) {
            const int gcol = col0 + wn * 64 + t * 16 + (lane & 15);
            const float bv = bias ? bias[gcol] : 0.f;
#pragma unroll
            for (int r = 0; r < 8; ++r) {
                const int grow = row0 + wm * 32 + m * 16 + r + rhi;
                float v = acc[m][t][r] * scale + bv;
                if (doGelu) v = 0.5f * v * (1.f + erff(v * 0.70710678118654752f));
                const long long ci = cbase + (long long)grow * ldc + gcol;
                if (outF) outF[ci] = v;
                if (outB) outB[ci] = f2bf(v);
            }
        }
    }
}

// ---------------------------------------------------------------------------
// Row softmax over S=2048 f32 scores; writes bf16 probs IN PLACE
// (row r: bf16 at ushort offset r*4096, i.e. same bytes as its f32 row).
// ---------------------------------------------------------------------------
__global__ __launch_bounds__(256) void softmax_rows(float* Sc) {
    __shared__ float red[256];
    const long long row = blockIdx.x;
    const float* rp = Sc + row * 2048ll;
    unsigned short* pp = (unsigned short*)Sc + row * 4096ll;
    const int tid = threadIdx.x;

    float v[8];
    float mx = -3.4e38f;
#pragma unroll
    for (int i = 0; i < 8; ++i) { v[i] = rp[tid + i * 256]; mx = fmaxf(mx, v[i]); }
    red[tid] = mx; __syncthreads();
    for (int s = 128; s > 0; s >>= 1) {
        if (tid < s) red[tid] = fmaxf(red[tid], red[tid + s]);
        __syncthreads();
    }
    mx = red[0]; __syncthreads();

    float sum = 0.f;
#pragma unroll
    for (int i = 0; i < 8; ++i) { v[i] = expf(v[i] - mx); sum += v[i]; }
    red[tid] = sum; __syncthreads();
    for (int s = 128; s > 0; s >>= 1) {
        if (tid < s) red[tid] += red[tid + s];
        __syncthreads();
    }
    const float inv = 1.f / red[0];
#pragma unroll
    for (int i = 0; i < 8; ++i) pp[tid + i * 256] = f2bf(v[i] * inv);
}

// ---------------------------------------------------------------------------
// out = LayerNorm(a + b) * g + beta   (one block per row, D=1024)
// Safe when outF aliases a (each thread rewrites only what it read).
// ---------------------------------------------------------------------------
__global__ __launch_bounds__(256) void add_ln(
    const float* a, const float* b, const float* __restrict__ g,
    const float* __restrict__ be, float* outF, unsigned short* outB)
{
    __shared__ float red[256];
    const long long base = (long long)blockIdx.x * 1024;
    const int tid = threadIdx.x;

    float v[4]; float s = 0.f;
#pragma unroll
    for (int i = 0; i < 4; ++i) {
        int c = tid + i * 256;
        v[i] = a[base + c] + b[base + c];
        s += v[i];
    }
    red[tid] = s; __syncthreads();
    for (int st = 128; st > 0; st >>= 1) {
        if (tid < st) red[tid] += red[tid + st];
        __syncthreads();
    }
    const float mean = red[0] * (1.f / 1024.f); __syncthreads();

    float vs = 0.f;
#pragma unroll
    for (int i = 0; i < 4; ++i) { float d = v[i] - mean; vs += d * d; }
    red[tid] = vs; __syncthreads();
    for (int st = 128; st > 0; st >>= 1) {
        if (tid < st) red[tid] += red[tid + st];
        __syncthreads();
    }
    const float rstd = rsqrtf(red[0] * (1.f / 1024.f) + 1e-5f);

#pragma unroll
    for (int i = 0; i < 4; ++i) {
        int c = tid + i * 256;
        float o = (v[i] - mean) * rstd * g[c] + be[c];
        if (outF) outF[base + c] = o;
        if (outB) outB[base + c] = f2bf(o);
    }
}

// ---------------------------------------------------------------------------
// Host orchestration
// ---------------------------------------------------------------------------
extern "C" void kernel_launch(void* const* d_in, const int* in_sizes, int n_in,
                              void* d_out, int out_size, void* d_ws, size_t ws_size,
                              hipStream_t stream) {
    (void)in_sizes; (void)n_in; (void)out_size; (void)ws_size;
    const int Bt = 4, S = 2048, D = 1024;
    const long long MS = (long long)Bt * S;            // 8192 rows

    const float* x    = (const float*)d_in[0];
    const float* Wq   = (const float*)d_in[1];
    const float* bq   = (const float*)d_in[2];
    const float* Wk   = (const float*)d_in[3];
    const float* bk   = (const float*)d_in[4];
    const float* Wv   = (const float*)d_in[5];
    const float* bv   = (const float*)d_in[6];
    const float* g_at = (const float*)d_in[7];
    const float* b_at = (const float*)d_in[8];
    const float* g_ln = (const float*)d_in[9];
    const float* b_ln = (const float*)d_in[10];
    const float* W1   = (const float*)d_in[11];
    const float* c1   = (const float*)d_in[12];
    const float* W2   = (const float*)d_in[13];
    const float* c2   = (const float*)d_in[14];
    float* out = (float*)d_out;

    // bump allocator over d_ws
    char* wsp = (char*)d_ws;
    size_t off = 0;
    auto alloc = [&](size_t bytes) -> void* {
        void* p = wsp + off;
        off = (off + bytes + 255) & ~(size_t)255;
        return p;
    };
    unsigned short* xb  = (unsigned short*)alloc(MS * D * 2);       // 16 MB
    unsigned short* Wqb = (unsigned short*)alloc((size_t)D * D * 2);
    unsigned short* Wkb = (unsigned short*)alloc((size_t)D * D * 2);
    unsigned short* Wvb = (unsigned short*)alloc((size_t)D * D * 2);
    unsigned short* W1b = (unsigned short*)alloc((size_t)D * D * 2);
    unsigned short* W2b = (unsigned short*)alloc((size_t)D * D * 2);
    unsigned short* Qb  = (unsigned short*)alloc(MS * D * 2);
    unsigned short* Kb  = (unsigned short*)alloc(MS * D * 2);
    unsigned short* Vb  = (unsigned short*)alloc(MS * D * 2);
    float* Sc   = (float*)alloc((size_t)Bt * S * S * 4);            // 67 MB (L2-resident)
    float* attn = (float*)alloc(MS * D * 4);                        // reused: attn -> at -> h(res)
    unsigned short* hb  = (unsigned short*)alloc(MS * D * 2);
    unsigned short* g1b = (unsigned short*)alloc(MS * D * 2);
    float* g2 = Sc;                                                 // reuse score buffer for FFN2 out

    // 1) convert activations + weights to bf16
    cvt_f32_bf16<<<4096, 256, 0, stream>>>(x,  xb,  MS * D);
    cvt_f32_bf16<<<2048, 256, 0, stream>>>(Wq, Wqb, (long long)D * D);
    cvt_f32_bf16<<<2048, 256, 0, stream>>>(Wk, Wkb, (long long)D * D);
    cvt_f32_bf16<<<2048, 256, 0, stream>>>(Wv, Wvb, (long long)D * D);
    cvt_f32_bf16<<<2048, 256, 0, stream>>>(W1, W1b, (long long)D * D);
    cvt_f32_bf16<<<2048, 256, 0, stream>>>(W2, W2b, (long long)D * D);

    dim3 blk(256);
    // 2) Q/K/V = x @ W + b  (M=8192,N=1024,K=1024) -> bf16
    dim3 gQKV(D / 128, (unsigned)(MS / 128), 1);
    wmma_gemm_bf16<0><<<gQKV, blk, 0, stream>>>(xb, Wqb, nullptr, Qb, bq,
        D, D, D, D, 0, 0, 0, 1.f, 0);
    wmma_gemm_bf16<0><<<gQKV, blk, 0, stream>>>(xb, Wkb, nullptr, Kb, bk,
        D, D, D, D, 0, 0, 0, 1.f, 0);
    wmma_gemm_bf16<0><<<gQKV, blk, 0, stream>>>(xb, Wvb, nullptr, Vb, bv,
        D, D, D, D, 0, 0, 0, 1.f, 0);

    // 3) scores = (Q @ K^T) / 8   per batch (M=N=2048, K=1024) -> f32 (fits L2)
    dim3 gSc(S / 128, S / 128, Bt);
    wmma_gemm_bf16<1><<<gSc, blk, 0, stream>>>(Qb, Kb, Sc, nullptr, nullptr,
        D, D, D, S,
        (long long)S * D, (long long)S * D, (long long)S * S,
        0.125f, 0);

    // 4) softmax rows, bf16 probs written in place (row ld = 4096 ushorts)
    softmax_rows<<<(unsigned)MS, 256, 0, stream>>>(Sc);

    // 5) attn = P @ V  per batch (M=2048,N=1024,K=2048)
    dim3 gAV(D / 128, S / 128, Bt);
    wmma_gemm_bf16<0><<<gAV, blk, 0, stream>>>((const unsigned short*)Sc, Vb,
        attn, nullptr, nullptr,
        S, 2 * S /*lda(P)*/, D, D,
        (long long)S * 2 * S, (long long)S * D, (long long)S * D,
        1.f, 0);

    // 6) at = LN(attn + x) ; 7) h = LN(at + x)  (both in place on attn)
    add_ln<<<(unsigned)MS, 256, 0, stream>>>(attn, x, g_at, b_at, attn, nullptr);
    add_ln<<<(unsigned)MS, 256, 0, stream>>>(attn, x, g_ln, b_ln, attn, hb);

    // 8) g1 = gelu(h @ W1 + c1) -> bf16 ; 9) g2 = gelu(g1 @ W2 + c2) -> f32
    wmma_gemm_bf16<0><<<gQKV, blk, 0, stream>>>(hb, W1b, nullptr, g1b, c1,
        D, D, D, D, 0, 0, 0, 1.f, 1);
    wmma_gemm_bf16<0><<<gQKV, blk, 0, stream>>>(g1b, W2b, g2, nullptr, c2,
        D, D, D, D, 0, 0, 0, 1.f, 1);

    // 10) out = LN(g2 + h_res)
    add_ln<<<(unsigned)MS, 256, 0, stream>>>(g2, attn, g_ln, b_ln, out, nullptr);
}
